// TnetConv_64106681860601
// MI455X (gfx1250) — matrix operand order repaired and checked by reading, not compiled, not run on hardware
//
#include <hip/hip_runtime.h>

// ---------- vector types ----------
typedef __bf16 bf16_t;
typedef bf16_t v16bf __attribute__((ext_vector_type(16)));
typedef float  v8f   __attribute__((ext_vector_type(8)));
typedef float  f32x4 __attribute__((ext_vector_type(4)));
typedef unsigned int u32x4 __attribute__((ext_vector_type(4)));
typedef unsigned short u16;

// ---------- geometry ----------
#define P_IMG   4096          // H*W
#define C_IN    256
#define NPIX    65536         // B*H*W
#define PXWG    256           // pixels per workgroup
#define NWG     (NPIX / PXWG) // 256 workgroups

// padded LDS row lengths (elements); all rows 16B-aligned
#define KPAD   40             // x panel: 32 ch + pad
#define CPAD1  264            // W1 row: 256 + 8
#define CPAD2  136            // W2 row: 128 + 8
#define CPAD3  72             // W3 row: 64 + 8
#define MPAD1  136            // h1 row: 128 + 8
#define MPAD2  72             // h2 row: 64 + 8

// ---------- LDS layout (byte offsets, all 16B aligned) ----------
#define XPAN_ELEMS (256 * KPAD)       // per buffer (u16)
#define OFF_XPAN 0                    // 2 buffers: 2*256*40*2 = 40960
#define OFF_W1   40960                // 128*264*2 = 67584
#define OFF_W2   108544               // 64*136*2  = 17408
#define OFF_W3   125952               // 16*72*2   = 2304
#define OFF_SA1  128256               // 128 f32
#define OFF_SB1  128768
#define OFF_SA2  129280               // 64 f32
#define OFF_SB2  129536
#define OFF_SA3  129792               // 16 f32
#define OFF_SB3  129856
#define OFF_H1   129920               // 8*32*136*2 = 69632
#define OFF_H2   199552               // 8*32*72*2  = 36864
#define OFF_H3   236416               // 8*32*16*4  = 16384
#define SMEM_BYTES 252800

union BfTile { u32x4 q[2]; v16bf v; };

static __device__ __forceinline__ u16 bf1(float a) {
  unsigned u = __float_as_uint(a);
  return (u16)((u + 0x7FFFu + ((u >> 16) & 1u)) >> 16);
}
// RNE round both floats to bf16, pack into one dword via v_perm_b32
static __device__ __forceinline__ unsigned pk2(float a, float b) {
  unsigned ua = __float_as_uint(a), ub = __float_as_uint(b);
  ua += 0x7FFFu + ((ua >> 16) & 1u);
  ub += 0x7FFFu + ((ub >> 16) & 1u);
  // dst bytes: [0]=ua.b2 [1]=ua.b3 [2]=ub.b2 [3]=ub.b3
  return __builtin_amdgcn_perm(ub, ua, 0x07060302u);
}
static __device__ __forceinline__ v16bf ld_tile(const u16* base, int e0, int e1) {
  BfTile tmp;
  tmp.q[0] = *reinterpret_cast<const u32x4*>(base + e0);
  tmp.q[1] = *reinterpret_cast<const u32x4*>(base + e1);
  return tmp.v;
}

// one cyclic Jacobi rotation on symmetric A, accumulating V (p,q constant after inline)
static __device__ __forceinline__ void jrot(float A[3][3], float V[3][3], int p, int q) {
  float apq = A[p][q];
  if (fabsf(apq) > 1e-20f) {
    float tau = (A[q][q] - A[p][p]) / (2.0f * apq);
    float tt  = copysignf(1.0f, tau) / (fabsf(tau) + sqrtf(1.0f + tau * tau));
    float c   = rsqrtf(1.0f + tt * tt);
    float s   = tt * c;
#pragma unroll
    for (int k = 0; k < 3; ++k) { float t1 = A[k][p], t2 = A[k][q]; A[k][p] = c*t1 - s*t2; A[k][q] = s*t1 + c*t2; }
#pragma unroll
    for (int k = 0; k < 3; ++k) { float t1 = A[p][k], t2 = A[q][k]; A[p][k] = c*t1 - s*t2; A[q][k] = s*t1 + c*t2; }
#pragma unroll
    for (int k = 0; k < 3; ++k) { float t1 = V[k][p], t2 = V[k][q]; V[k][p] = c*t1 - s*t2; V[k][q] = s*t1 + c*t2; }
  }
}

__global__ __launch_bounds__(256)
void tnet_fused(const float* __restrict__ x,
                const float* __restrict__ w1, const float* __restrict__ b1,
                const float* __restrict__ g1, const float* __restrict__ be1,
                const float* __restrict__ m1, const float* __restrict__ v1,
                const float* __restrict__ w2, const float* __restrict__ b2,
                const float* __restrict__ g2, const float* __restrict__ be2,
                const float* __restrict__ m2, const float* __restrict__ v2,
                const float* __restrict__ w3, const float* __restrict__ b3,
                const float* __restrict__ g3, const float* __restrict__ be3,
                const float* __restrict__ m3, const float* __restrict__ v3,
                float* __restrict__ out) {
  extern __shared__ char smem[];
  u16*   xpan = (u16*)(smem + OFF_XPAN);
  u16*   w1s  = (u16*)(smem + OFF_W1);
  u16*   w2s  = (u16*)(smem + OFF_W2);
  u16*   w3s  = (u16*)(smem + OFF_W3);
  float* sA1  = (float*)(smem + OFF_SA1);
  float* sB1  = (float*)(smem + OFF_SB1);
  float* sA2  = (float*)(smem + OFF_SA2);
  float* sB2  = (float*)(smem + OFF_SB2);
  float* sA3  = (float*)(smem + OFF_SA3);
  float* sB3  = (float*)(smem + OFF_SB3);
  u16*   h1s  = (u16*)(smem + OFF_H1);
  u16*   h2s  = (u16*)(smem + OFF_H2);
  float* h3s  = (float*)(smem + OFF_H3);

  const int t     = threadIdx.x;
  const int wave  = t >> 5;
  const int lane  = t & 31;
  const int lg    = lane >> 4;   // lane-group (K/N half select)
  const int l16   = lane & 15;   // row within half
  const int g0    = blockIdx.x * PXWG;           // global pixel base
  const int bimg  = g0 >> 12;                    // /4096 (PXWG divides P_IMG)
  const int p0    = g0 & (P_IMG - 1);
  const float* xb = x + (size_t)bimg * C_IN * P_IMG + p0;

  // ---------------- one-time weight / BN staging ----------------
  for (int i = t * 4; i < 128 * 256; i += 1024) {
    f32x4 v = *reinterpret_cast<const f32x4*>(w1 + i);
    int o = i >> 8, c = i & 255;
    unsigned q0 = pk2(v.x, v.y), q1 = pk2(v.z, v.w);
    unsigned* d = reinterpret_cast<unsigned*>(w1s + o * CPAD1 + c);
    d[0] = q0; d[1] = q1;
  }
  for (int i = t * 4; i < 64 * 128; i += 1024) {
    f32x4 v = *reinterpret_cast<const f32x4*>(w2 + i);
    int o = i >> 7, c = i & 127;
    unsigned q0 = pk2(v.x, v.y), q1 = pk2(v.z, v.w);
    unsigned* d = reinterpret_cast<unsigned*>(w2s + o * CPAD2 + c);
    d[0] = q0; d[1] = q1;
  }
  for (int i = t; i < 16 * 64; i += 256) {
    int o = i >> 6, c = i & 63;
    float v = (o < 9) ? w3[o * 64 + c] : 0.0f;
    w3s[o * CPAD3 + c] = bf1(v);
  }
  if (t < 128) { float sc = g1[t] / sqrtf(v1[t] + 1e-5f); sA1[t] = sc; sB1[t] = fmaf(b1[t] - m1[t], sc, be1[t]); }
  if (t < 64)  { float sc = g2[t] / sqrtf(v2[t] + 1e-5f); sA2[t] = sc; sB2[t] = fmaf(b2[t] - m2[t], sc, be2[t]); }
  if (t < 16)  {
    if (t < 9) { float sc = g3[t] / sqrtf(v3[t] + 1e-5f); sA3[t] = sc; sB3[t] = fmaf(b3[t] - m3[t], sc, be3[t]); }
    else       { sA3[t] = 0.0f; sB3[t] = 0.0f; }
  }

  // cooperative x-panel stage: 32 channels x 256 pixels, f32 -> bf16, [px][ch].
  // Each thread handles a pair of adjacent channels so LDS stores are packed b32.
  auto stageX = [&](int kt, int buf) {
    u16* pan = xpan + buf * XPAN_ELEMS;
    const float* base = xb + (size_t)kt * 32 * P_IMG;
    f32x4 va[4], vb[4];
#pragma unroll
    for (int j = 0; j < 4; ++j) {
      int pi = j * 256 + t;
      int c2 = (pi >> 6) * 2;       // even channel of the pair
      int px = (pi & 63) * 4;       // pixel group of 4
      const float* s0 = base + (size_t)c2 * P_IMG + px;
      va[j] = *reinterpret_cast<const f32x4*>(s0);
      vb[j] = *reinterpret_cast<const f32x4*>(s0 + P_IMG);
      if (kt + 1 < 8) __builtin_prefetch(s0 + 32 * P_IMG, 0, 0);
    }
#pragma unroll
    for (int j = 0; j < 4; ++j) {
      int pi = j * 256 + t;
      int c2 = (pi >> 6) * 2;
      int px = (pi & 63) * 4;
#pragma unroll
      for (int u = 0; u < 4; ++u) {
        unsigned q = pk2(va[j][u], vb[j][u]);   // (c2, c2+1) packed
        *reinterpret_cast<unsigned*>(pan + (px + u) * KPAD + c2) = q;
      }
    }
  };

  // ---------------- layer 1: 128 <- 256, K-panel double buffered ----------------
  v8f acc1[8][2] = {};
  stageX(0, 0);
  for (int kt = 0; kt < 8; ++kt) {
    __syncthreads();
    if (kt + 1 < 8) stageX(kt + 1, (kt + 1) & 1);
    const u16* pan = xpan + (kt & 1) * XPAN_ELEMS;
    v16bf bt[2];
#pragma unroll
    for (int nt = 0; nt < 2; ++nt) {
      int pxl = wave * 32 + nt * 16 + l16;
      bt[nt] = ld_tile(pan, pxl * KPAD + 8 * lg, pxl * KPAD + 16 + 8 * lg);
    }
    // software-pipelined A-tile loads: load A(mt+1) before WMMA(mt)
    int ra0 = (0 * 16 + l16) * CPAD1 + kt * 32;
    v16bf at = ld_tile(w1s, ra0 + 8 * lg, ra0 + 16 + 8 * lg);
#pragma unroll
    for (int mt = 0; mt < 8; ++mt) {
      v16bf atn;
      if (mt < 7) {
        int ra = ((mt + 1) * 16 + l16) * CPAD1 + kt * 32;
        atn = ld_tile(w1s, ra + 8 * lg, ra + 16 + 8 * lg);
      }
      acc1[mt][0] = __builtin_amdgcn_wmma_f32_16x16x32_bf16(
          false, at, false, bt[0], (short)0, acc1[mt][0], false, false);
      acc1[mt][1] = __builtin_amdgcn_wmma_f32_16x16x32_bf16(
          false, at, false, bt[1], (short)0, acc1[mt][1], false, false);
      if (mt < 7) at = atn;
    }
  }

  // BN + ReLU + repack -> h1 [px][ch] bf16 (per-wave region)
  u16* h1w = h1s + wave * 32 * MPAD1;
#pragma unroll
  for (int mt = 0; mt < 8; ++mt)
#pragma unroll
    for (int nt = 0; nt < 2; ++nt) {
      int pxw = nt * 16 + l16;
      int c0  = mt * 16 + 8 * lg;
      float vv[8];
#pragma unroll
      for (int r = 0; r < 8; ++r)
        vv[r] = fmaxf(fmaf(acc1[mt][nt][r], sA1[c0 + r], sB1[c0 + r]), 0.0f);
      u32x4 q = { pk2(vv[0], vv[1]), pk2(vv[2], vv[3]), pk2(vv[4], vv[5]), pk2(vv[6], vv[7]) };
      *reinterpret_cast<u32x4*>(h1w + pxw * MPAD1 + c0) = q;
    }
  __syncthreads();

  // ---------------- layer 2: 64 <- 128 ----------------
  v8f acc2[4][2] = {};
#pragma unroll
  for (int kt = 0; kt < 4; ++kt) {
    int k0 = kt * 32;
    v16bf bt[2];
#pragma unroll
    for (int nt = 0; nt < 2; ++nt) {
      int pxw = nt * 16 + l16;
      bt[nt] = ld_tile(h1w, pxw * MPAD1 + k0 + 8 * lg, pxw * MPAD1 + k0 + 16 + 8 * lg);
    }
    int ra0 = l16 * CPAD2 + k0;
    v16bf at = ld_tile(w2s, ra0 + 8 * lg, ra0 + 16 + 8 * lg);
#pragma unroll
    for (int mt = 0; mt < 4; ++mt) {
      v16bf atn;
      if (mt < 3) {
        int ra = ((mt + 1) * 16 + l16) * CPAD2 + k0;
        atn = ld_tile(w2s, ra + 8 * lg, ra + 16 + 8 * lg);
      }
      acc2[mt][0] = __builtin_amdgcn_wmma_f32_16x16x32_bf16(
          false, at, false, bt[0], (short)0, acc2[mt][0], false, false);
      acc2[mt][1] = __builtin_amdgcn_wmma_f32_16x16x32_bf16(
          false, at, false, bt[1], (short)0, acc2[mt][1], false, false);
      if (mt < 3) at = atn;
    }
  }
  u16* h2w = h2s + wave * 32 * MPAD2;
#pragma unroll
  for (int mt = 0; mt < 4; ++mt)
#pragma unroll
    for (int nt = 0; nt < 2; ++nt) {
      int pxw = nt * 16 + l16;
      int c0  = mt * 16 + 8 * lg;
      float vv[8];
#pragma unroll
      for (int r = 0; r < 8; ++r)
        vv[r] = fmaxf(fmaf(acc2[mt][nt][r], sA2[c0 + r], sB2[c0 + r]), 0.0f);
      u32x4 q = { pk2(vv[0], vv[1]), pk2(vv[2], vv[3]), pk2(vv[4], vv[5]), pk2(vv[6], vv[7]) };
      *reinterpret_cast<u32x4*>(h2w + pxw * MPAD2 + c0) = q;
    }
  __syncthreads();

  // ---------------- layer 3: 16 (9 valid) <- 64 ----------------
  v8f acc3[2] = {};
#pragma unroll
  for (int kt = 0; kt < 2; ++kt) {
    int k0 = kt * 32;
    int ra = l16 * CPAD3 + k0;
    v16bf at = ld_tile(w3s, ra + 8 * lg, ra + 16 + 8 * lg);
#pragma unroll
    for (int nt = 0; nt < 2; ++nt) {
      int pxw = nt * 16 + l16;
      v16bf bt = ld_tile(h2w, pxw * MPAD2 + k0 + 8 * lg, pxw * MPAD2 + k0 + 16 + 8 * lg);
      acc3[nt] = __builtin_amdgcn_wmma_f32_16x16x32_bf16(
          false, at, false, bt, (short)0, acc3[nt], false, false);
    }
  }
  float* h3w = h3s + wave * 32 * 16;
#pragma unroll
  for (int nt = 0; nt < 2; ++nt) {
    int pxw = nt * 16 + l16;
    int c0  = 8 * lg;
    float vv[8];
#pragma unroll
    for (int r = 0; r < 8; ++r)
      vv[r] = fmaxf(fmaf(acc3[nt][r], sA3[c0 + r], sB3[c0 + r]), 0.0f);
    f32x4 qa = { vv[0], vv[1], vv[2], vv[3] };
    f32x4 qb = { vv[4], vv[5], vv[6], vv[7] };
    *reinterpret_cast<f32x4*>(h3w + pxw * 16 + c0)     = qa;
    *reinterpret_cast<f32x4*>(h3w + pxw * 16 + c0 + 4) = qb;
  }
  __syncthreads();

  // ---------------- Procrustes: each lane owns one pixel's 3x3 ----------------
  const float* mp = h3w + lane * 16;
  float M[3][3];
#pragma unroll
  for (int i = 0; i < 3; ++i)
#pragma unroll
    for (int j = 0; j < 3; ++j) M[i][j] = mp[3 * i + j];

  float detM = M[0][0] * (M[1][1] * M[2][2] - M[1][2] * M[2][1])
             - M[0][1] * (M[1][0] * M[2][2] - M[1][2] * M[2][0])
             + M[0][2] * (M[1][0] * M[2][1] - M[1][1] * M[2][0]);

  float A[3][3], V[3][3];
#pragma unroll
  for (int i = 0; i < 3; ++i)
#pragma unroll
    for (int j = 0; j < 3; ++j) {
      float s = 0.0f;
#pragma unroll
      for (int k = 0; k < 3; ++k) s = fmaf(M[k][i], M[k][j], s);
      A[i][j] = s;
      V[i][j] = (i == j) ? 1.0f : 0.0f;
    }
#pragma unroll
  for (int sweep = 0; sweep < 5; ++sweep) {
    jrot(A, V, 0, 1); jrot(A, V, 0, 2); jrot(A, V, 1, 2);
  }
  float lam[3] = { A[0][0], A[1][1], A[2][2] };
  int imin = 0;
  if (lam[1] < lam[imin]) imin = 1;
  if (lam[2] < lam[imin]) imin = 2;
  float sg = (detM < 0.0f) ? -1.0f : 1.0f;
  float dinv[3];
#pragma unroll
  for (int i = 0; i < 3; ++i) {
    float sv = sqrtf(fmaxf(lam[i], 0.0f));
    float f  = (i == imin) ? sg : 1.0f;
    dinv[i]  = f / fmaxf(sv, 1e-8f);
  }
  float S[3][3];
#pragma unroll
  for (int i = 0; i < 3; ++i)
#pragma unroll
    for (int j = 0; j < 3; ++j) {
      float s = 0.0f;
#pragma unroll
      for (int k = 0; k < 3; ++k) s = fmaf(V[i][k] * dinv[k], V[j][k], s);
      S[i][j] = s;
    }
  float* op = out + (size_t)(g0 + wave * 32 + lane) * 9;
#pragma unroll
  for (int i = 0; i < 3; ++i)
#pragma unroll
    for (int j = 0; j < 3; ++j) {
      float s = 0.0f;
#pragma unroll
      for (int k = 0; k < 3; ++k) s = fmaf(M[i][k], S[k][j], s);
      op[3 * i + j] = s;
    }
}

extern "C" void kernel_launch(void* const* d_in, const int* in_sizes, int n_in,
                              void* d_out, int out_size, void* d_ws, size_t ws_size,
                              hipStream_t stream) {
  (void)in_sizes; (void)n_in; (void)out_size; (void)d_ws; (void)ws_size;
  const float* x   = (const float*)d_in[0];
  const float* w1  = (const float*)d_in[1];
  const float* b1  = (const float*)d_in[2];
  const float* g1  = (const float*)d_in[3];
  const float* be1 = (const float*)d_in[4];
  const float* m1  = (const float*)d_in[5];
  const float* v1  = (const float*)d_in[6];
  const float* w2  = (const float*)d_in[7];
  const float* b2  = (const float*)d_in[8];
  const float* g2  = (const float*)d_in[9];
  const float* be2 = (const float*)d_in[10];
  const float* m2  = (const float*)d_in[11];
  const float* v2  = (const float*)d_in[12];
  const float* w3  = (const float*)d_in[13];
  const float* b3  = (const float*)d_in[14];
  const float* g3  = (const float*)d_in[15];
  const float* be3 = (const float*)d_in[16];
  const float* m3  = (const float*)d_in[17];
  const float* v3  = (const float*)d_in[18];
  float* out = (float*)d_out;

  hipFuncSetAttribute((const void*)tnet_fused,
                      hipFuncAttributeMaxDynamicSharedMemorySize, SMEM_BYTES);
  tnet_fused<<<dim3(NWG), dim3(256), SMEM_BYTES, stream>>>(
      x, w1, b1, g1, be1, m1, v1,
      w2, b2, g2, be2, m2, v2,
      w3, b3, g3, be3, m3, v3, out);
}
